// MultiHeadAttn_5463198401106
// MI455X (gfx1250) — compile-verified, matrix operand order checked by reading, not run on hardware
//
#include <hip/hip_runtime.h>
#include <hip/hip_bf16.h>

// ---------------- problem constants (from reference) ----------------
#define BZ      2
#define SEQ     2048
#define D_MODEL 1024
#define N_HEADS 16
#define DEPTH   64          // D_MODEL / N_HEADS

typedef __attribute__((ext_vector_type(16))) __bf16 v16bf;
typedef __attribute__((ext_vector_type(8)))  float  v8f;
typedef __attribute__((ext_vector_type(4))) unsigned int u32x4;
typedef __attribute__((ext_vector_type(8))) int         i32x8;
typedef __attribute__((ext_vector_type(4))) int         i32x4;

// pointer types matching the async-to-LDS builtin prototype
typedef int v4i __attribute__((ext_vector_type(4)));
typedef __attribute__((address_space(1))) v4i* v4i_gp;   // global
typedef __attribute__((address_space(3))) v4i* v4i_lp;   // LDS

// ---- CDNA5 data-movement feature detection (device pass only) ----
#if defined(__gfx1250__)
#if __has_builtin(__builtin_amdgcn_global_load_async_to_lds_b128)
#define HAVE_ASYNC_LDS 1
#endif
#if __has_builtin(__builtin_amdgcn_tensor_load_to_lds)
#define HAVE_TDM 1
#endif
#endif

// 16-byte global -> LDS copy: async (no VGPR transit, ASYNCcnt) when available
__device__ __forceinline__ void cp16(void* lds, const void* g) {
#if defined(HAVE_ASYNC_LDS)
  __builtin_amdgcn_global_load_async_to_lds_b128((v4i_gp)g, (v4i_lp)lds, 0, 0);
#else
  *(uint4*)lds = *(const uint4*)g;
#endif
}
// wait until at most N async ops remain in flight (no-op on fallback path:
// manual ds_store staging is ordered by the barrier's DScnt wait)
__device__ __forceinline__ void async_wait0() {
#if defined(HAVE_ASYNC_LDS)
  asm volatile("s_wait_asynccnt 0x0" ::: "memory");
#endif
}
__device__ __forceinline__ void async_wait2() {
#if defined(HAVE_ASYNC_LDS)
  asm volatile("s_wait_asynccnt 0x2" ::: "memory");
#endif
}
__device__ __forceinline__ void async_wait4() {
#if defined(HAVE_ASYNC_LDS)
  asm volatile("s_wait_asynccnt 0x4" ::: "memory");
#endif
}

// -----------------------------------------------------------------------
// bf16 WMMA fragment loaders (CDNA5 ISA 7.12.2 16-bit A/B layout).
// Per-lane element i maps to K(i) = 16*(i>>3) + 8*half + (i&7).
// -----------------------------------------------------------------------
__device__ __forceinline__ v16bf ld_frag_row(const __bf16* p, int half) {
  v16bf r;
#pragma unroll
  for (int i = 0; i < 16; ++i)
    r[i] = p[16 * (i >> 3) + 8 * half + (i & 7)];
  return r;
}

__device__ __forceinline__ v16bf ld_frag_col(const __bf16* base, int col, int ld, int half) {
  v16bf r;
#pragma unroll
  for (int i = 0; i < 16; ++i) {
    int K = 16 * (i >> 3) + 8 * half + (i & 7);
    r[i] = base[K * ld + col];
  }
  return r;
}

// ---------------- fp32 -> bf16 convert ----------------
__global__ __launch_bounds__(256) void cvt_f32_bf16(const float* __restrict__ x,
                                                    __bf16* __restrict__ y, int n) {
  int i = (blockIdx.x * 256 + threadIdx.x) * 4;
  if (i + 3 < n) {
    float4 f = *(const float4*)(x + i);
    y[i + 0] = (__bf16)f.x;
    y[i + 1] = (__bf16)f.y;
    y[i + 2] = (__bf16)f.z;
    y[i + 3] = (__bf16)f.w;
  }
}

// -----------------------------------------------------------------------
// Y[M,N] = X[M,K] @ W[N,K]^T (nn.Linear), bf16 in, bf16/f32 out.
// 64x64 tile per 256-thread block; double-buffered async LDS staging so the
// DMA for tile k+1 overlaps the WMMAs of tile k (s_wait_asynccnt 2).
// -----------------------------------------------------------------------
template <bool OUT_F32>
__global__ __launch_bounds__(256) void gemm_xwT(const __bf16* __restrict__ X,
                                                const __bf16* __restrict__ W,
                                                void* __restrict__ Y,
                                                int M, int N, int K) {
  __shared__ __bf16 As[2][64][40];   // 80B rows: 16B-aligned, conflict-free
  __shared__ __bf16 Bs[2][64][40];

  const int tid  = threadIdx.x;
  const int lane = tid & 31, wave = tid >> 5;
  const int half = lane >> 4, l16 = lane & 15;
  const int wr = wave & 3, wc = wave >> 2;
  const int m0 = blockIdx.x * 64, n0 = blockIdx.y * 64;

  const int sr = tid >> 2;          // staging row 0..63
  const int sc = (tid & 3) * 8;     // staging col 0/8/16/24

  v8f acc0 = {};
  v8f acc1 = {};

  // prologue: stage tile 0 into buffer 0
  cp16(&As[0][sr][sc], X + (size_t)(m0 + sr) * K + sc);
  cp16(&Bs[0][sr][sc], W + (size_t)(n0 + sr) * K + sc);

  for (int k0 = 0, it = 0; k0 < K; k0 += 32, ++it) {
    const int cur = it & 1, nxt = cur ^ 1;
    const bool more = (k0 + 32 < K);
    if (more) {  // stage tile k+1 while tile k computes
      const __bf16* xp = X + (size_t)(m0 + sr) * K + k0 + 32 + sc;
      const __bf16* wp = W + (size_t)(n0 + sr) * K + k0 + 32 + sc;
      cp16(&As[nxt][sr][sc], xp);
      cp16(&Bs[nxt][sr][sc], wp);
      __builtin_prefetch(xp + 32, 0, 1);     // -> global_prefetch_b8 (k+2)
      __builtin_prefetch(wp + 32, 0, 1);
      async_wait2();                         // drain only tile k's 2 copies
    } else {
      async_wait0();
    }
    __syncthreads();

    v16bf a  = ld_frag_row(&As[cur][wr * 16 + l16][0], half);
    v16bf b0 = ld_frag_row(&Bs[cur][wc * 32 + l16][0], half);
    v16bf b1 = ld_frag_row(&Bs[cur][wc * 32 + 16 + l16][0], half);
    acc0 = __builtin_amdgcn_wmma_f32_16x16x32_bf16(false, a, false, b0, (short)0, acc0, false, false);
    acc1 = __builtin_amdgcn_wmma_f32_16x16x32_bf16(false, a, false, b1, (short)0, acc1, false, false);
    __syncthreads();   // all waves done reading 'cur' before it is re-staged
  }

#pragma unroll
  for (int r = 0; r < 8; ++r) {
    int m = m0 + wr * 16 + 8 * half + r;     // C/D: M = r + 8*half, N = lane&15
    int n = n0 + wc * 32 + l16;
    if (OUT_F32) {
      ((float*)Y)[(size_t)m * N + n]      = acc0[r];
      ((float*)Y)[(size_t)m * N + n + 16] = acc1[r];
    } else {
      ((__bf16*)Y)[(size_t)m * N + n]      = (__bf16)acc0[r];
      ((__bf16*)Y)[(size_t)m * N + n + 16] = (__bf16)acc1[r];
    }
  }
}

// -----------------------------------------------------------------------
// TDM: one descriptor moves a 128x64 bf16 tile, hardware-padding each 128B
// row with 16B (32+4 DWORDs -> LDS stride 72 elements).
// -----------------------------------------------------------------------
#if defined(HAVE_TDM)
__device__ __forceinline__ void tdm_load_k128(const __bf16* gsrc, __bf16* ldsdst) {
  unsigned long long ga = (unsigned long long)(uintptr_t)gsrc;
  u32x4 g0;
  g0[0] = 1u;                                            // count=1
  g0[1] = (unsigned)(uintptr_t)ldsdst;                   // lds_addr
  g0[2] = (unsigned)ga;                                  // global_addr[31:0]
  g0[3] = (unsigned)((ga >> 32) & 0x01FFFFFFu) | (2u << 30);  // addr[56:32] | type=2
  i32x8 g1;
  g1[0] = (1 << 16) | (1 << 20) | (4 << 22) | (3 << 25); // data_size=2B, pad_en, ivl=32dw, amt=4dw
  g1[1] = (DEPTH << 16);                                 // tensor_dim0 = 64
  g1[2] = (int)((unsigned)SEQ << 16);                    // tensor_dim1 = 2048
  g1[3] = (DEPTH << 16);                                 // tile_dim0 = 64
  g1[4] = 128;                                           // tile_dim1 = 128
  g1[5] = DEPTH;                                         // tensor_dim0_stride = 64
  g1[6] = 0;
  g1[7] = 0;
  i32x4 gz = {0, 0, 0, 0};
#if __clang_major__ >= 23
  i32x8 gz8 = {0, 0, 0, 0, 0, 0, 0, 0};
  __builtin_amdgcn_tensor_load_to_lds(g0, g1, gz, gz, gz8, 0);
#else
  __builtin_amdgcn_tensor_load_to_lds(g0, g1, gz, gz, 0);
#endif
}
#endif

// -----------------------------------------------------------------------
// att[b,h,q,:] = softmax( sqrt(Qh Kh^T) + msk*(-1e9) )
// One block per (b, h, 16 q-rows); K streamed in double-buffered 128-row
// chunks via the Tensor Data Mover (chunk n+1 in flight during chunk n).
// -----------------------------------------------------------------------
__global__ __launch_bounds__(256) void attn_scores_softmax(const __bf16* __restrict__ qp,
                                                           const __bf16* __restrict__ kp,
                                                           const float* __restrict__ msk,
                                                           float* __restrict__ att) {
  extern __shared__ char smem[];
  const int SROW_LD = SEQ + 4;                               // stride%64==4: conflict-free
  float*  srow = (float*)smem;                               // 16 * 2052 * 4B
  __bf16* Qs   = (__bf16*)(smem + 16 * SROW_LD * 4);         // 16 * 72 * 2B
  __bf16* Ks   = (__bf16*)((char*)Qs + 16 * 72 * 2);         // 2 * 128 * 72 * 2B

  const int tid  = threadIdx.x;
  const int lane = tid & 31, wave = tid >> 5;
  const int half = lane >> 4, l16 = lane & 15;
  const int qb = blockIdx.x * 16, h = blockIdx.y, b = blockIdx.z;

  // raw-reshape head split: head block is contiguous [SEQ, DEPTH]
  const __bf16* Qh = qp + (size_t)b * SEQ * D_MODEL + (size_t)h * SEQ * DEPTH;
  const __bf16* Kh = kp + (size_t)b * SEQ * D_MODEL + (size_t)h * SEQ * DEPTH;

  if (tid < 128) {  // stage 16x64 Q tile (16B chunks)
    int r = tid >> 3, c = (tid & 7) * 8;
    cp16(&Qs[r * 72 + c], Qh + (size_t)(qb + r) * DEPTH + c);
  }
  async_wait0();

  // prologue: first K chunk into buffer 0
#if defined(HAVE_TDM)
  if (tid < 32) tdm_load_k128(Kh, Ks);
#else
#pragma unroll
  for (int j = 0; j < 4; ++j) {
    int ch = tid + 256 * j;
    int r = ch >> 3, c = (ch & 7) * 8;
    cp16(&Ks[r * 72 + c], Kh + (size_t)r * DEPTH + c);
  }
#endif

  for (int kb = 0, it = 0; kb < SEQ; kb += 128, ++it) {
    const int cur = it & 1, nxt = cur ^ 1;
    const bool more = (kb + 128 < SEQ);
    __bf16* KsC = Ks + cur * (128 * 72);
    __bf16* KsN = Ks + nxt * (128 * 72);

#if defined(HAVE_TDM)
    if (tid < 32) {
      if (more) {
        tdm_load_k128(Kh + (size_t)(kb + 128) * DEPTH, KsN);
        __builtin_amdgcn_s_wait_tensorcnt(1);   // drain only chunk k's DMA
      } else {
        __builtin_amdgcn_s_wait_tensorcnt(0);
      }
    }
#else
    if (more) {
#pragma unroll
      for (int j = 0; j < 4; ++j) {
        int ch = tid + 256 * j;
        int r = ch >> 3, c = (ch & 7) * 8;
        cp16(&KsN[r * 72 + c], Kh + (size_t)(kb + 128 + r) * DEPTH + c);
      }
      async_wait4();
    } else {
      async_wait0();
    }
#endif
    __syncthreads();

    v8f acc = {};
#pragma unroll
    for (int d = 0; d < DEPTH; d += 32) {
      v16bf a  = ld_frag_row(&Qs[l16 * 72 + d], half);
      v16bf bk = ld_frag_row(&KsC[(wave * 16 + l16) * 72 + d], half);
      acc = __builtin_amdgcn_wmma_f32_16x16x32_bf16(false, a, false, bk, (short)0, acc, false, false);
    }

    int col = kb + wave * 16 + l16;
    float mv = msk[b * SEQ + col] * -1e9f;
#pragma unroll
    for (int r = 0; r < 8; ++r) {
      float s = fmaxf(acc[r], 0.0f);
      srow[(r + 8 * half) * SROW_LD + col] = __builtin_amdgcn_sqrtf(s) + mv;
    }
    __syncthreads();   // reads of 'cur' done before it is re-staged
  }

  // softmax: 16 threads per row; wave32 butterfly reductions in 16-lane halves
  const int row = tid >> 4, sub = tid & 15;
  float* sr = srow + row * SROW_LD;

  float mx = -1e30f;
  for (int j = sub; j < SEQ; j += 16) mx = fmaxf(mx, sr[j]);
#pragma unroll
  for (int m = 8; m >= 1; m >>= 1) mx = fmaxf(mx, __shfl_xor(mx, m, 32));

  float sum = 0.0f;
  for (int j = sub; j < SEQ; j += 16) {
    float e = __builtin_amdgcn_exp2f((sr[j] - mx) * 1.44269504088896f);
    sr[j] = e;
    sum += e;
  }
#pragma unroll
  for (int m = 8; m >= 1; m >>= 1) sum += __shfl_xor(sum, m, 32);
  float inv = __builtin_amdgcn_rcpf(sum);

  float* arow = att + (((size_t)b * N_HEADS + h) * SEQ + (qb + row)) * (size_t)SEQ;
  for (int j = sub; j < SEQ; j += 16) arow[j] = sr[j] * inv;
}

// -----------------------------------------------------------------------
// ctx[b,h,q,d] = sum_k att[b,h,q,k] * V[b,h,k,d]   (att fp32 -> bf16 on stage)
// V tile DMA'd async while the fp32->bf16 conversion runs on VALU.
// -----------------------------------------------------------------------
__global__ __launch_bounds__(256) void ctx_gemm(const float* __restrict__ att,
                                                const __bf16* __restrict__ vp,
                                                __bf16* __restrict__ ctx) {
  __shared__ __bf16 As[64][40];
  __shared__ __bf16 Vs[32][72];

  const int tid  = threadIdx.x;
  const int lane = tid & 31, wave = tid >> 5;
  const int half = lane >> 4, l16 = lane & 15;
  const int wr = wave & 3, wc = wave >> 2;
  const int q0 = blockIdx.x * 64, h = blockIdx.y, b = blockIdx.z;

  const float*  Ah = att + (((size_t)b * N_HEADS + h) * SEQ) * (size_t)SEQ;
  const __bf16* Vh = vp  + (size_t)b * SEQ * D_MODEL + (size_t)h * SEQ * DEPTH;
  __bf16*       Ch = ctx + (size_t)b * SEQ * D_MODEL + (size_t)h * SEQ * DEPTH;

  const int sr = tid >> 2, sc = (tid & 3) * 8;   // att staging 64x32
  const int vr = tid >> 3, vc = (tid & 7) * 8;   // V staging 32x64

  v8f acc0 = {};
  v8f acc1 = {};

  for (int k0 = 0; k0 < SEQ; k0 += 32) {
    cp16(&Vs[vr][vc], Vh + (size_t)(k0 + vr) * DEPTH + vc);
    const float* ap = Ah + (size_t)(q0 + sr) * SEQ + k0 + sc;
    float4 f0 = *(const float4*)(ap);
    float4 f1 = *(const float4*)(ap + 4);
    __bf16* as = &As[sr][sc];
    as[0] = (__bf16)f0.x; as[1] = (__bf16)f0.y; as[2] = (__bf16)f0.z; as[3] = (__bf16)f0.w;
    as[4] = (__bf16)f1.x; as[5] = (__bf16)f1.y; as[6] = (__bf16)f1.z; as[7] = (__bf16)f1.w;
    async_wait0();
    __syncthreads();

    v16bf a  = ld_frag_row(&As[wr * 16 + l16][0], half);
    v16bf b0 = ld_frag_col(&Vs[0][0], wc * 32 + l16,      72, half);
    v16bf b1 = ld_frag_col(&Vs[0][0], wc * 32 + 16 + l16, 72, half);
    acc0 = __builtin_amdgcn_wmma_f32_16x16x32_bf16(false, a, false, b0, (short)0, acc0, false, false);
    acc1 = __builtin_amdgcn_wmma_f32_16x16x32_bf16(false, a, false, b1, (short)0, acc1, false, false);
    __syncthreads();
  }

#pragma unroll
  for (int r = 0; r < 8; ++r) {
    int q = q0 + wr * 16 + 8 * half + r;
    int n = wc * 32 + l16;
    Ch[(size_t)q * DEPTH + n]      = (__bf16)acc0[r];
    Ch[(size_t)q * DEPTH + n + 16] = (__bf16)acc1[r];
  }
}

// -----------------------------------------------------------------------
extern "C" void kernel_launch(void* const* d_in, const int* in_sizes, int n_in,
                              void* d_out, int out_size, void* d_ws, size_t ws_size,
                              hipStream_t stream) {
  (void)in_sizes; (void)n_in; (void)out_size; (void)ws_size;

  const float* v   = (const float*)d_in[0];
  const float* k   = (const float*)d_in[1];
  const float* q   = (const float*)d_in[2];
  const float* msk = (const float*)d_in[3];
  const float* Wq  = (const float*)d_in[4];
  const float* Wk  = (const float*)d_in[5];
  const float* Wv  = (const float*)d_in[6];
  const float* Wfc = (const float*)d_in[7];

  const size_t ND = (size_t)BZ * SEQ * D_MODEL;   // 4M activations
  const size_t NW = (size_t)D_MODEL * D_MODEL;    // 1M weights

  char* ws = (char*)d_ws;
  __bf16* qb   = (__bf16*)ws; ws += ND * 2;
  __bf16* kb   = (__bf16*)ws; ws += ND * 2;
  __bf16* vb   = (__bf16*)ws; ws += ND * 2;
  __bf16* Wqb  = (__bf16*)ws; ws += NW * 2;
  __bf16* Wkb  = (__bf16*)ws; ws += NW * 2;
  __bf16* Wvb  = (__bf16*)ws; ws += NW * 2;
  __bf16* Wfcb = (__bf16*)ws; ws += NW * 2;
  __bf16* qp   = (__bf16*)ws; ws += ND * 2;
  __bf16* kp   = (__bf16*)ws; ws += ND * 2;
  __bf16* vpb  = (__bf16*)ws; ws += ND * 2;
  __bf16* ctxb = (__bf16*)ws; ws += ND * 2;       // ~64 MB total

  float* outp = (float*)d_out;                    // [BZ*SEQ, D_MODEL] fp32
  float* attp = outp + ND;                        // [BZ, H, SEQ, SEQ] fp32

  // 1) convert inputs/weights to bf16
  cvt_f32_bf16<<<(int)(ND / 1024), 256, 0, stream>>>(q, qb, (int)ND);
  cvt_f32_bf16<<<(int)(ND / 1024), 256, 0, stream>>>(k, kb, (int)ND);
  cvt_f32_bf16<<<(int)(ND / 1024), 256, 0, stream>>>(v, vb, (int)ND);
  cvt_f32_bf16<<<(int)(NW / 1024), 256, 0, stream>>>(Wq, Wqb, (int)NW);
  cvt_f32_bf16<<<(int)(NW / 1024), 256, 0, stream>>>(Wk, Wkb, (int)NW);
  cvt_f32_bf16<<<(int)(NW / 1024), 256, 0, stream>>>(Wv, Wvb, (int)NW);
  cvt_f32_bf16<<<(int)(NW / 1024), 256, 0, stream>>>(Wfc, Wfcb, (int)NW);

  // 2) QKV projections: [4096,1024] @ [1024,1024]^T
  dim3 gp(64, 16);
  gemm_xwT<false><<<gp, 256, 0, stream>>>(qb, Wqb, qp,  BZ * SEQ, D_MODEL, D_MODEL);
  gemm_xwT<false><<<gp, 256, 0, stream>>>(kb, Wkb, kp,  BZ * SEQ, D_MODEL, D_MODEL);
  gemm_xwT<false><<<gp, 256, 0, stream>>>(vb, Wvb, vpb, BZ * SEQ, D_MODEL, D_MODEL);

  // 3) scores + softmax -> att (fp32, second half of d_out)
  int smem = 16 * (SEQ + 4) * 4 + 16 * 72 * 2 + 2 * 128 * 72 * 2;   // ~167 KB dynamic LDS
  (void)hipFuncSetAttribute((const void*)attn_scores_softmax,
                            hipFuncAttributeMaxDynamicSharedMemorySize, smem);
  attn_scores_softmax<<<dim3(SEQ / 16, N_HEADS, BZ), 256, smem, stream>>>(qp, kp, msk, attp);

  // 4) ctx = att @ V
  ctx_gemm<<<dim3(SEQ / 64, N_HEADS, BZ), 256, 0, stream>>>(attp, vpb, ctxb);

  // 5) out = ctx @ Wfc^T  (fp32 into first part of d_out)
  gemm_xwT<true><<<gp, 256, 0, stream>>>(ctxb, Wfcb, outp, BZ * SEQ, D_MODEL, D_MODEL);
}
